// HamiltonianLayer_21723944583172
// MI455X (gfx1250) — compile-verified, matrix operand order checked by reading, not run on hardware
//
#include <hip/hip_runtime.h>
#include <hip/hip_bf16.h>

// Problem constants (match reference)
#define BATCH 8
#define SEQ   1024
#define EMBED 512
#define NREL  4

typedef __attribute__((ext_vector_type(16))) __bf16 v16bf;
typedef __attribute__((ext_vector_type(2)))  __bf16 v2bf;
typedef __attribute__((ext_vector_type(8)))  float  v8f;

// Pack two floats into two packed bf16.
//  - If the target declares v_cvt_pk_bf16_f32, use it (true RNE, 1 instr).
//  - Else pack by truncation with a single v_perm_b32: selector 0x07060302
//    picks {hi[31:16], lo[31:16]} -> (hi & 0xFFFF0000) | (lo >> 16).
//    (<=1 ulp of bf16 vs RNE; negligible against bf16 quantization.)
__device__ __forceinline__ unsigned int pack2bf(float lo, float hi) {
#if __has_builtin(__builtin_amdgcn_cvt_pk_bf16_f32)
    union { v2bf v; unsigned int u; } c;
    c.v = __builtin_amdgcn_cvt_pk_bf16_f32(lo, hi);
    return c.u;
#else
    return __builtin_amdgcn_perm(__float_as_uint(hi), __float_as_uint(lo),
                                 0x07060302u);
#endif
}

// Build a 16-bit A/B WMMA fragment from an LDS row of 32 bf16 values.
// Per CDNA5 ISA 7.12.2: lane<16 holds K in {0..7, 16..23}, lane>=16 holds
// K in {8..15, 24..31}; two contiguous 16-byte runs -> two ds_load_b128.
__device__ __forceinline__ v16bf frag_from_lds(const unsigned short* rowp, int half) {
    union { v16bf v; uint4 q[2]; } u;
    u.q[0] = *(const uint4*)(rowp + half * 8);
    u.q[1] = *(const uint4*)(rowp + 16 + half * 8);
    return u.v;
}

// Kernel 1: h[r][k] = sum_l H[r][k][l]   (R*E = 2048 rows of 512 floats)
__global__ void HamiltonianLayer_hsum_kernel(const float* __restrict__ H,
                                             float* __restrict__ h) {
    int idx = blockIdx.x * blockDim.x + threadIdx.x;
    if (idx >= NREL * EMBED) return;
    const float* row = H + (size_t)idx * EMBED;
    float s = 0.0f;
    for (int l = 0; l < EMBED; ++l) s += row[l];
    h[idx] = s;
}

// Kernel 2: energy[b,i,j] = sum_k x[b,i,k] * h[rel[b,i,j],k] * x[b,j,k]
// 256 threads = 8 waves; per-WG output tile 128(M) x 64(N); K-tile 32.
// A tile (128x32) is unscaled x_i; B tiles (4 x 64x32) carry diag(h_r)*x_j.
// LDS double-buffered; global loads for tile k+1 overlap WMMAs for tile k.
__global__ __launch_bounds__(256)
void HamiltonianLayer_energy_kernel(const float* __restrict__ X,
                                    const int*   __restrict__ RI,
                                    const float* __restrict__ Hs,   // [NREL][EMBED]
                                    float*       __restrict__ Out) {
    __shared__ float hs[NREL][EMBED];                               // 8 KB
    __shared__ __align__(16) unsigned short As[2][128][32];         // 16 KB
    __shared__ __align__(16) unsigned short Bs[2][NREL][64][32];    // 32 KB

    const int tid    = threadIdx.x;
    const int lane   = tid & 31;
    const int wave   = tid >> 5;
    const int wm     = wave & 3;      // wave M quadrant (0..3) -> rows wm*32..+32
    const int wn     = wave >> 2;     // wave N quadrant (0..1) -> cols wn*32..+32
    const int lanelo = lane & 15;
    const int half   = lane >> 4;

    const int b  = blockIdx.z;
    const int m0 = blockIdx.y * 128;
    const int n0 = blockIdx.x * 64;

    // Prefetch the relation-index tile for the epilogue (32 KB, 256 lines):
    // thread t prefetches line (row = t>>1, half-row = t&1) -> global_prefetch.
    {
        const size_t roff = (size_t)(b * SEQ + m0 + (tid >> 1)) * SEQ + n0 + (tid & 1) * 32;
        __builtin_prefetch(RI + roff, 0, 1);
    }

    // Stage the whole reduced hamiltonian vector table once.
    for (int idx = tid; idx < NREL * EMBED; idx += 256)
        hs[idx >> 9][idx & (EMBED - 1)] = Hs[idx];

    // Staging decomposition: A = 128 rows x 8 float4-quads (1024 work items,
    // 4 per thread); B = 64 rows x 8 quads (512 work items, 2 per thread).
    const int arow[4] = { (tid + 0 * 256) >> 3, (tid + 1 * 256) >> 3,
                          (tid + 2 * 256) >> 3, (tid + 3 * 256) >> 3 };
    const int aq      = tid & 7;
    const int brow[2] = { (tid + 0 * 256) >> 3, (tid + 1 * 256) >> 3 };

    float4 areg[4];
    float4 breg[2];

    auto load_regs = [&](int k0) {
#pragma unroll
        for (int it = 0; it < 4; ++it)
            areg[it] = *(const float4*)&X[(size_t)(b * SEQ + m0 + arow[it]) * EMBED + k0 + aq * 4];
#pragma unroll
        for (int it = 0; it < 2; ++it)
            breg[it] = *(const float4*)&X[(size_t)(b * SEQ + n0 + brow[it]) * EMBED + k0 + aq * 4];
    };

    auto store_lds = [&](int buf, int k0) {
        unsigned int* AsW = (unsigned int*)&As[buf][0][0];   // [128][16] pairs
#pragma unroll
        for (int it = 0; it < 4; ++it) {
            const int idx = (tid + it * 256);                 // == arow*8 + aq
            AsW[idx * 2 + 0] = pack2bf(areg[it].x, areg[it].y);
            AsW[idx * 2 + 1] = pack2bf(areg[it].z, areg[it].w);
        }
#pragma unroll
        for (int it = 0; it < 2; ++it) {
            const float4 v = breg[it];
            const int idx = (tid + it * 256);                 // == brow*8 + aq
#pragma unroll
            for (int r = 0; r < NREL; ++r) {
                const float4 hv = *(const float4*)&hs[r][k0 + aq * 4];
                unsigned int* BsW = (unsigned int*)&Bs[buf][r][0][0];
                BsW[idx * 2 + 0] = pack2bf(v.x * hv.x, v.y * hv.y);
                BsW[idx * 2 + 1] = pack2bf(v.z * hv.z, v.w * hv.w);
            }
        }
    };

    const v8f zero = {0.f, 0.f, 0.f, 0.f, 0.f, 0.f, 0.f, 0.f};
    v8f acc[NREL][4];
#pragma unroll
    for (int r = 0; r < NREL; ++r)
#pragma unroll
        for (int t = 0; t < 4; ++t) acc[r][t] = zero;

    load_regs(0);
    __syncthreads();   // hs table ready for all threads

    const int NT = EMBED / 32;
    for (int kt = 0; kt < NT; ++kt) {
        const int buf = kt & 1;
        store_lds(buf, kt * 32);
        __syncthreads();                       // LDS tile ready; prev buf free

        if (kt + 1 < NT) load_regs((kt + 1) * 32);   // overlap with WMMAs below

        v16bf afrag[2];
#pragma unroll
        for (int mm = 0; mm < 2; ++mm)
            afrag[mm] = frag_from_lds(&As[buf][wm * 32 + mm * 16 + lanelo][0], half);

#pragma unroll
        for (int r = 0; r < NREL; ++r) {
            v16bf bfrag[2];
#pragma unroll
            for (int nn = 0; nn < 2; ++nn)
                bfrag[nn] = frag_from_lds(&Bs[buf][r][wn * 32 + nn * 16 + lanelo][0], half);
#pragma unroll
            for (int mm = 0; mm < 2; ++mm)
#pragma unroll
                for (int nn = 0; nn < 2; ++nn)
                    acc[r][mm * 2 + nn] = __builtin_amdgcn_wmma_f32_16x16x32_bf16(
                        /*neg_a=*/false, afrag[mm],
                        /*neg_b=*/false, bfrag[nn],
                        /*c_mod=*/(short)0, acc[r][mm * 2 + nn],
                        /*reuse_a=*/false, /*reuse_b=*/false);
        }
    }

    // Epilogue: C layout (16x16 f32): VGPR v -> M = v + 8*half, N = lanelo.
    // Branch-free 2-level mux on the 2-bit relation index (no exec divergence).
#pragma unroll
    for (int mm = 0; mm < 2; ++mm)
#pragma unroll
        for (int nn = 0; nn < 2; ++nn) {
            const int t = mm * 2 + nn;
            const int i_base = m0 + wm * 32 + mm * 16 + half * 8;
            const int j      = n0 + wn * 32 + nn * 16 + lanelo;
            const size_t base = (size_t)(b * SEQ + i_base) * SEQ + j;
#pragma unroll
            for (int v = 0; v < 8; ++v) {
                const size_t off = base + (size_t)v * SEQ;
                const unsigned int rel = (unsigned int)RI[off];
                const unsigned int msk0 = 0u - (rel & 1u);
                const unsigned int msk1 = 0u - ((rel >> 1) & 1u);
                const unsigned int b0 = __float_as_uint(acc[0][t][v]);
                const unsigned int b1 = __float_as_uint(acc[1][t][v]);
                const unsigned int b2 = __float_as_uint(acc[2][t][v]);
                const unsigned int b3 = __float_as_uint(acc[3][t][v]);
                const unsigned int lo = (b0 & ~msk0) | (b1 & msk0);
                const unsigned int hi = (b2 & ~msk0) | (b3 & msk0);
                const unsigned int rs = (lo & ~msk1) | (hi & msk1);
                Out[off] = __uint_as_float(rs);
            }
        }
}

extern "C" void kernel_launch(void* const* d_in, const int* in_sizes, int n_in,
                              void* d_out, int out_size, void* d_ws, size_t ws_size,
                              hipStream_t stream) {
    // setup_inputs() order: token_embeddings (f32), relation_indices (int),
    // hamiltonians (f32). Output: energy (f32, B*S*S).
    const float* X   = (const float*)d_in[0];
    const int*   RI  = (const int*)d_in[1];    // integer -> const int* per harness contract
    const float* H   = (const float*)d_in[2];
    float*       Out = (float*)d_out;

    float* h = (float*)d_ws;                   // NREL*EMBED floats = 8 KB scratch

    HamiltonianLayer_hsum_kernel<<<(NREL * EMBED + 255) / 256, 256, 0, stream>>>(H, h);

    dim3 grid(SEQ / 64, SEQ / 128, BATCH);     // (N tiles, M tiles, batch)
    HamiltonianLayer_energy_kernel<<<grid, 256, 0, stream>>>(X, RI, h, Out);
}
